// MADE_85753317032699
// MI455X (gfx1250) — matrix operand order repaired and checked by reading, not compiled, not run on hardware
//
#include <hip/hip_runtime.h>
#include <hip/hip_bf16.h>
#include <stdint.h>

// MADE autoregressive sampling, D=32, H=1024, L=2, B=4096.
// Masks folded into bf16 weights once per launch; 32 sequential steps of
// [3 x WMMA bf16 GEMM (bias+ReLU fused) -> 2-column output step].
// GEMM staging uses CDNA5 async global->LDS copies (ASYNCcnt) with
// double-buffered LDS when the builtins are available.

typedef __attribute__((ext_vector_type(16))) __bf16 v16bf;
typedef __attribute__((ext_vector_type(8)))  __bf16 v8bf;
typedef __attribute__((ext_vector_type(8)))  float  v8f;
typedef __attribute__((ext_vector_type(4)))  int    v4i;

static constexpr int kD = 32;
static constexpr int kH = 1024;
static constexpr int kB = 4096;

#if defined(__has_builtin)
#if __has_builtin(__builtin_amdgcn_global_load_async_to_lds_b128) && \
    __has_builtin(__builtin_amdgcn_s_wait_asynccnt)
#define USE_ASYNC_LDS 1
#endif
#endif

typedef __attribute__((address_space(1))) v4i GlobalV4;   // global int4
typedef __attribute__((address_space(3))) v4i LocalV4;    // LDS int4

__device__ __forceinline__ void copy16_to_lds(__bf16* lds, const __bf16* g) {
#if USE_ASYNC_LDS
    __builtin_amdgcn_global_load_async_to_lds_b128((GlobalV4*)g,
                                                   (LocalV4*)lds, 0, 0);
#else
    *(v8bf*)lds = *(const v8bf*)g;
#endif
}

// ------------------------- mask-folding prep kernels -------------------------
__global__ void prep_w0(const float* __restrict__ W0, __bf16* __restrict__ W0m) {
    int i = blockIdx.x * blockDim.x + threadIdx.x;       // H*D = 32768
    if (i >= kH * kD) return;
    int j = i / kD, k = i % kD;
    float m = ((j % (kD - 1)) >= k) ? 1.0f : 0.0f;       // d_hid >= d_in
    W0m[i] = (__bf16)(m * W0[i]);
}

__global__ void prep_wh(const float* __restrict__ Wh, __bf16* __restrict__ Whm) {
    int i = blockIdx.x * blockDim.x + threadIdx.x;       // L*H*H = 2097152
    if (i >= 2 * kH * kH) return;
    int jk = i % (kH * kH);
    int j = jk / kH, k = jk % kH;
    float m = ((j % (kD - 1)) >= (k % (kD - 1))) ? 1.0f : 0.0f;
    Whm[i] = (__bf16)(m * Wh[i]);
}

__global__ void prep_wo(const float* __restrict__ Wout, float* __restrict__ Wom) {
    int i = blockIdx.x * blockDim.x + threadIdx.x;       // 2D*H = 65536
    if (i >= 2 * kD * kH) return;
    int o = i / kH, k = i % kH;
    int deg_out = (o % kD) - 1;                          // output degree
    float m = (deg_out >= (int)(k % (kD - 1))) ? 1.0f : 0.0f;
    Wom[i] = m * Wout[i];
}

__global__ void zero_u32(uint32_t* __restrict__ p, int n) {
    int i = blockIdx.x * blockDim.x + threadIdx.x;
    if (i < n) p[i] = 0u;
}

// ------------------------- WMMA bf16 GEMM + bias + ReLU ----------------------
// Out[M,1024] = relu( A[M,K] * W[1024,K]^T + bias ), bf16 in/out, f32 acc.
// 256 threads = 8 waves in 4(M) x 2(N); wave tile 32M x 64N; block 128M x 128N.
// K staged 32 at a time through double-buffered LDS via async global->LDS.
__global__ __launch_bounds__(256) void made_gemm(
    const __bf16* __restrict__ A, int lda, int K,
    const __bf16* __restrict__ W,              // [1024, K] row-major
    const float*  __restrict__ bias,           // [1024]
    __bf16*       __restrict__ Out)            // [M, 1024]
{
    __shared__ __bf16 As[2][128][40];          // +8 pad: conflict-free b128 reads
    __shared__ __bf16 Bs[2][128][40];

    const int t     = threadIdx.x;
    const int bn    = blockIdx.x;              // N tile (of 8)
    const int bm    = blockIdx.y;              // M tile (of 32)
    const int wave  = t >> 5, lane = t & 31;
    const int half  = lane >> 4, l16 = lane & 15;
    const int mbase = (wave >> 1) * 32;
    const int nbase = (wave & 1) * 64;
    const int Arow0 = bm * 128, Brow0 = bn * 128;

    v8f acc[2][4];
#pragma unroll
    for (int mf = 0; mf < 2; ++mf)
#pragma unroll
        for (int nf = 0; nf < 4; ++nf)
#pragma unroll
            for (int r = 0; r < 8; ++r) acc[mf][nf][r] = 0.0f;

    // Each thread stages 2 x 16B chunks of A and 2 x 16B of B per K-step
    // (4 async instructions per wave per buffer).
    auto issue = [&](int buf, int kt) {
#pragma unroll
        for (int i = 0; i < 2; ++i) {
            int id  = t * 2 + i;               // 0..511
            int row = id >> 2, c = (id & 3) << 3;
            copy16_to_lds(&As[buf][row][c],
                          A + (size_t)(Arow0 + row) * lda + kt + c);
            copy16_to_lds(&Bs[buf][row][c],
                          W + (size_t)(Brow0 + row) * K + kt + c);
        }
    };

    const int niter = K >> 5;
    issue(0, 0);
    for (int it = 0; it < niter; ++it) {
        const int cur = it & 1;
        if (it + 1 < niter) {
            issue(cur ^ 1, (it + 1) << 5);
#if USE_ASYNC_LDS
            __builtin_amdgcn_s_wait_asynccnt(4);   // cur's 4 done, next in flight
#endif
        } else {
#if USE_ASYNC_LDS
            __builtin_amdgcn_s_wait_asynccnt(0);
#endif
        }
        __syncthreads();

        union U { v16bf v; v8bf h[2]; };
        // A fragments (16x32 MxK): lane<16 -> K {0..7,16..23}; lane>=16 -> +8
        v16bf af[2];
#pragma unroll
        for (int mf = 0; mf < 2; ++mf) {
            U u;
            int row = mbase + 16 * mf + l16;
            u.h[0] = *(const v8bf*)&As[cur][row][half * 8];
            u.h[1] = *(const v8bf*)&As[cur][row][16 + half * 8];
            af[mf] = u.v;
        }
        // B fragments (32x16 KxN): lane<16 -> K {0..7,16..23}; lane>=16 -> +8
        v16bf bfr[4];
#pragma unroll
        for (int nf = 0; nf < 4; ++nf) {
            U u;
            int n = nbase + 16 * nf + l16;
            u.h[0] = *(const v8bf*)&Bs[cur][n][half * 8];
            u.h[1] = *(const v8bf*)&Bs[cur][n][16 + half * 8];
            bfr[nf] = u.v;
        }
#pragma unroll
        for (int mf = 0; mf < 2; ++mf)
#pragma unroll
            for (int nf = 0; nf < 4; ++nf)
                acc[mf][nf] = __builtin_amdgcn_wmma_f32_16x16x32_bf16(
                    false, af[mf], false, bfr[nf], (short)0, acc[mf][nf],
                    false, false);
        __syncthreads();   // protect buffer reuse two iterations ahead
    }

    // Epilogue: C layout -> VGPR r: lanes0-15 (M=r,N=lane); lanes16-31 (M=r+8).
#pragma unroll
    for (int mf = 0; mf < 2; ++mf)
#pragma unroll
        for (int nf = 0; nf < 4; ++nf) {
            int col  = Brow0 + nbase + 16 * nf + l16;
            float bv = bias[col];
#pragma unroll
            for (int r = 0; r < 8; ++r) {
                int row = Arow0 + mbase + 16 * mf + r + 8 * half;
                float v = acc[mf][nf][r] + bv;
                v = fmaxf(v, 0.0f);
                Out[(size_t)row * kH + col] = (__bf16)v;
            }
        }
}

// -------------------- per-step output columns + AR update --------------------
// One wave per batch row: mu = Wom[idx].a + bout[idx], ls = Wom[idx+32].a + ...
// Writes the new column to d_out (f32) AND to the persistent bf16 state xbf.
__global__ __launch_bounds__(256) void made_out_step(
    const __bf16* __restrict__ a, const float* __restrict__ Wom,
    const float* __restrict__ bout, const float* __restrict__ z,
    float* __restrict__ xout, __bf16* __restrict__ xbf, int idx)
{
    int gid  = blockIdx.x * blockDim.x + threadIdx.x;
    int row  = gid >> 5, lane = gid & 31;
    const __bf16* ar = a + (size_t)row * kH;
    const float* w0 = Wom + (size_t)idx * kH;
    const float* w1 = Wom + (size_t)(idx + kD) * kH;
    float mu = 0.0f, ls = 0.0f;
    for (int k = lane; k < kH; k += 32) {
        float av = (float)ar[k];
        mu += av * w0[k];
        ls += av * w1[k];
    }
#pragma unroll
    for (int off = 16; off > 0; off >>= 1) {
        mu += __shfl_xor(mu, off, 32);
        ls += __shfl_xor(ls, off, 32);
    }
    if (lane == 0) {
        mu += bout[idx];
        ls += bout[idx + kD];
        float xi = z[(size_t)row * kD + idx] * __expf(ls) + mu;
        xout[(size_t)row * kD + idx] = xi;
        xbf [(size_t)row * kD + idx] = (__bf16)xi;
    }
}

// --------------------------------- launcher ----------------------------------
extern "C" void kernel_launch(void* const* d_in, const int* in_sizes, int n_in,
                              void* d_out, int out_size, void* d_ws, size_t ws_size,
                              hipStream_t stream) {
    (void)in_sizes; (void)n_in; (void)out_size; (void)ws_size;
    const float* z    = (const float*)d_in[0];
    const float* W0   = (const float*)d_in[1];
    const float* b0   = (const float*)d_in[2];
    const float* Wh   = (const float*)d_in[3];
    const float* bh   = (const float*)d_in[4];
    const float* Wout = (const float*)d_in[5];
    const float* bout = (const float*)d_in[6];

    // Workspace layout (~21.6 MB total)
    uint8_t* ws = (uint8_t*)d_ws;
    __bf16* W0m = (__bf16*)(ws + 0);          //   65536 B: [1024,32]   bf16
    __bf16* Whm = (__bf16*)(ws + 65536);      // 4194304 B: [2,1024,1024] bf16
    float*  Wom = (float*) (ws + 4259840);    //  262144 B: [64,1024]   f32
    __bf16* xbf = (__bf16*)(ws + 4521984);    //  262144 B: [4096,32]   bf16
    __bf16* a0  = (__bf16*)(ws + 4784128);    // 8388608 B: [4096,1024] bf16
    __bf16* a1  = (__bf16*)(ws + 13172736);   // 8388608 B: [4096,1024] bf16

    // Fold MADE masks into weights (L2-resident afterwards); zero AR state.
    prep_w0<<<128, 256, 0, stream>>>(W0, W0m);
    prep_wh<<<8192, 256, 0, stream>>>(Wh, Whm);
    prep_wo<<<256, 256, 0, stream>>>(Wout, Wom);
    zero_u32<<<256, 256, 0, stream>>>((uint32_t*)xbf, kB * kD / 2);

    dim3 gg(kH / 128, kB / 128);   // (8, 32) blocks of 256 threads
    for (int idx = 0; idx < kD; ++idx) {
        made_gemm<<<gg, 256, 0, stream>>>(xbf, kD, kD,  W0m,           b0,      a0);
        made_gemm<<<gg, 256, 0, stream>>>(a0,  kH, kH,  Whm,           bh,      a1);
        made_gemm<<<gg, 256, 0, stream>>>(a1,  kH, kH,  Whm + kH * kH, bh + kH, a0);
        made_out_step<<<512, 256, 0, stream>>>(a0, Wom, bout, z,
                                               (float*)d_out, xbf, idx);
    }
}
